// DriveTargetProposer_32195074851241
// MI455X (gfx1250) — matrix operand order repaired and checked by reading, not compiled, not run on hardware
//
#include <hip/hip_runtime.h>

// ---------- problem constants ----------
#define NUM_DRIVES   64
#define SENSORY_DIM  256
#define LOCATION_DIM 16
#define MEM_SIZE     200000
#define NUM_RESULTS  5
#define THRESHOLD    0.1f
#define DIV_LAMBDA   0.5f
#define NEGV        -1000000000.0f
#define EPSV         1e-8f

// Tiles: N = 200000 = 12500 * 16 ; M = 64 = 4 * 16 ; K = 256 = 8 * 32 (exact)
#define NTILES 12500
#define GEMM_BLOCKS (NTILES / 4)   // 4 waves per block, one ntile per wave

typedef __attribute__((ext_vector_type(16))) __bf16 v16bf;
typedef __attribute__((ext_vector_type(8)))  __bf16 v8bf;
typedef __attribute__((ext_vector_type(8)))  float  v8f;
typedef __attribute__((ext_vector_type(4)))  int    v4i;

typedef unsigned long long u64;
typedef unsigned int       u32;

#define AS_GLOBAL __attribute__((address_space(1)))
#define AS_LDS    __attribute__((address_space(3)))

// ---------- workspace layout (bytes, all 256B aligned) ----------
#define WS_SN       0ull                          // 200000*256*2 = 102,400,000
#define WS_QN       102400000ull                  // 64*256*2 = 32768
#define WS_SEL      102432768ull                  // 64*256*2 = 32768
#define WS_SIMS     102465536ull                  // 200000*64*4 = 51,200,000 ([n][q])
#define WS_PEN      153665536ull                  // 200000*64*4
#define WS_SLOTS    204865536ull                  // 64 * u64
#define WS_COUNTS   204866048ull                  // 64 * int
#define WS_SELIDX   204866304ull                  // 5*64 * int

// ---------- output layout (floats) ----------
#define OUT_LOC    0
#define OUT_SDS    5120      // 64*5*16
#define OUT_SENSE  10240     // + 64*5*16
#define OUT_NFOUND 92160     // + 64*5*256

// =====================================================================
// Stage the 64x256 bf16 A matrix (32 KB) into LDS.
// Prefer the CDNA5 async-to-LDS path (ASYNCcnt), fall back to plain copy.
// Caller must __syncthreads() afterwards.
// =====================================================================
__device__ __forceinline__ void stage_A_lds(const unsigned short* __restrict__ Au,
                                            unsigned short* ldsA, int tid)
{
#if __has_builtin(__builtin_amdgcn_global_load_async_to_lds_b128)
    #pragma unroll
    for (int i = 0; i < 16; ++i) {
        __builtin_amdgcn_global_load_async_to_lds_b128(
            (AS_GLOBAL v4i*)(Au + tid * 8 + i * 1024),
            (AS_LDS    v4i*)(ldsA + tid * 8 + i * 1024),
            0, 0);
    }
  #if __has_builtin(__builtin_amdgcn_s_wait_asynccnt)
    __builtin_amdgcn_s_wait_asynccnt(0);
  #else
    asm volatile("s_wait_asynccnt 0x0" ::: "memory");
  #endif
#else
    #pragma unroll
    for (int i = 0; i < 16; ++i)
        *(v8bf*)(ldsA + tid * 8 + i * 1024) =
            *(const v8bf*)(Au + tid * 8 + i * 1024);
#endif
}

// =====================================================================
// WMMA core: one wave computes D[16n x 64q] for its ntile from LDS-A
// and streamed global B.  acc[mt] holds M-tile mt.
// =====================================================================
__device__ __forceinline__ void wmma_tile(const unsigned short* ldsA,
                                          const __bf16* __restrict__ SN,
                                          int srow, int nsub, int half,
                                          v8f acc[4])
{
    const __bf16* brow = SN + (size_t)srow * SENSORY_DIM + half * 8;
    #pragma unroll
    for (int k0 = 0; k0 < SENSORY_DIM; k0 += 32) {
        // B tile 32x16: lane holds K chunks [c..c+7],[c+16..c+23], c=half*8
        v8bf blo = *(const v8bf*)(brow + k0);
        v8bf bhi = *(const v8bf*)(brow + k0 + 16);
        v16bf b = __builtin_shufflevector(blo, bhi,
                    0,1,2,3,4,5,6,7,8,9,10,11,12,13,14,15);
        #pragma unroll
        for (int mt = 0; mt < 4; ++mt) {
            // A tile 16x32: lane holds K chunks [c..c+7],[c+16..c+23], c=half*8
            const unsigned short* ap =
                ldsA + (mt * 16 + nsub) * SENSORY_DIM + half * 8 + k0;
            v8bf lo = *(const v8bf*)ap;
            v8bf hi = *(const v8bf*)(ap + 16);
            v16bf a = __builtin_shufflevector(lo, hi,
                        0,1,2,3,4,5,6,7,8,9,10,11,12,13,14,15);
            acc[mt] = __builtin_amdgcn_wmma_f32_16x16x32_bf16(
                        false, a, false, b, (short)0, acc[mt], false, false);
        }
    }
}

__device__ __forceinline__ u32 float_key(float f) {
    u32 b = __float_as_uint(f);
    return (b & 0x80000000u) ? ~b : (b | 0x80000000u);
}

// =====================================================================
// 1) Normalize rows of a [nrows x 256] f32 matrix -> bf16 (one wave/row)
// =====================================================================
__global__ __launch_bounds__(256) void normalize_bf16_kernel(
    const float* __restrict__ src, unsigned short* __restrict__ dst, int nrows)
{
    int lane = threadIdx.x & 31;
    int wave = threadIdx.x >> 5;
    int row  = blockIdx.x * 8 + wave;
    if (row >= nrows) return;

    const float* s = src + (size_t)row * SENSORY_DIM + lane * 8;
    float v[8];
    #pragma unroll
    for (int i = 0; i < 8; ++i) v[i] = s[i];

    float ss = 0.f;
    #pragma unroll
    for (int i = 0; i < 8; ++i) ss += v[i] * v[i];
    #pragma unroll
    for (int off = 16; off > 0; off >>= 1) ss += __shfl_xor(ss, off, 32);

    float inv = 1.0f / fmaxf(sqrtf(ss), EPSV);

    v8bf o;
    #pragma unroll
    for (int i = 0; i < 8; ++i) o[i] = (__bf16)(v[i] * inv);
    *(v8bf*)((__bf16*)dst + (size_t)row * SENSORY_DIM + lane * 8) = o;
}

// =====================================================================
// 2) sims GEMM: simsT[n][64] = (qn . sn^T)^T via WMMA, A staged in LDS
// =====================================================================
__global__ __launch_bounds__(128) void gemm_sims_kernel(
    const unsigned short* __restrict__ Au,   // 64x256 bf16 (normalized queries)
    const unsigned short* __restrict__ SNu,  // 200000x256 bf16 (normalized)
    float* __restrict__ out)                 // [200000][64]
{
    __shared__ unsigned short ldsA[NUM_DRIVES * SENSORY_DIM];
    int tid = threadIdx.x;
    stage_A_lds(Au, ldsA, tid);
    __syncthreads();

    int lane  = tid & 31;
    int wave  = tid >> 5;
    int ntile = blockIdx.x * 4 + wave;
    int nsub  = lane & 15;
    int half  = lane >> 4;
    int srow  = ntile * 16 + nsub;

    v8f acc[4];
    v8f zero = {0.f,0.f,0.f,0.f,0.f,0.f,0.f,0.f};
    #pragma unroll
    for (int mt = 0; mt < 4; ++mt) acc[mt] = zero;

    wmma_tile(ldsA, (const __bf16*)SNu, srow, nsub, half, acc);

    // D layout: lane -> N = lane&15, M = r + 8*(lane>>4); store transposed.
    int mbase = 8 * half;
    #pragma unroll
    for (int mt = 0; mt < 4; ++mt)
        *(v8f*)(out + (size_t)srow * NUM_DRIVES + mt * 16 + mbase) = acc[mt];
}

// =====================================================================
// 3) Step 0: streaming argmax over masked sims + valid count
// =====================================================================
__global__ __launch_bounds__(256) void pass0_kernel(
    const float* __restrict__ simsT, u64* __restrict__ slots,
    int* __restrict__ counts)
{
    __shared__ u64 sbest[NUM_DRIVES];
    __shared__ int scnt[NUM_DRIVES];

    int tid = threadIdx.x;
    int q   = tid & 63;
    int g   = tid >> 6;
    if (tid < NUM_DRIVES) { sbest[tid] = 0ull; scnt[tid] = 0; }
    __syncthreads();

    int base = blockIdx.x * 128;
    u64 best = 0ull;
    int cnt  = 0;
    #pragma unroll 4
    for (int i = 0; i < 32; ++i) {
        int n = base + g + 4 * i;
        if (n < MEM_SIZE) {
            float s = simsT[(size_t)n * NUM_DRIVES + q];
            bool valid = (s >= THRESHOLD);
            cnt += valid ? 1 : 0;
            float adj = valid ? s : NEGV;
            u64 pk = ((u64)float_key(adj) << 32) | (u32)(~(u32)n);
            best = best > pk ? best : pk;
        }
    }
    atomicMax(&sbest[q], best);
    if (cnt) atomicAdd(&scnt[q], cnt);
    __syncthreads();
    if (tid < NUM_DRIVES) {
        atomicMax(&slots[tid], sbest[tid]);
        if (scnt[tid]) atomicAdd(&counts[tid], scnt[tid]);
    }
}

// =====================================================================
// 4) Fused step t=1..4: newsim = sel . sn^T (WMMA), penalty max-update,
//    adj = masked(sims) - lambda*penalty, selected masking, block argmax.
//    Saves a full re-read of penalty+sims vs separate GEMM + pass.
// =====================================================================
__global__ __launch_bounds__(128) void fused_step_kernel(
    const unsigned short* __restrict__ Au,   // selected senses of step t-1 (bf16)
    const unsigned short* __restrict__ SNu,
    float* __restrict__ penalty,             // [n][64]
    const float* __restrict__ simsT,         // [n][64]
    const int* __restrict__ selected,        // [5][64]
    u64* __restrict__ slots, int t)
{
    __shared__ unsigned short ldsA[NUM_DRIVES * SENSORY_DIM];
    __shared__ u64 sbest[NUM_DRIVES];
    __shared__ v4i ssel4[NUM_DRIVES];        // per-q selected indices (padded -1)

    int tid = threadIdx.x;
    stage_A_lds(Au, ldsA, tid);
    if (tid < NUM_DRIVES) {
        sbest[tid] = 0ull;
        v4i s;
        #pragma unroll
        for (int j = 0; j < 4; ++j)
            s[j] = (j < t) ? selected[j * 64 + tid] : -1;
        ssel4[tid] = s;
    }
    __syncthreads();

    int lane  = tid & 31;
    int wave  = tid >> 5;
    int ntile = blockIdx.x * 4 + wave;
    int nsub  = lane & 15;
    int half  = lane >> 4;
    int srow  = ntile * 16 + nsub;           // this lane's memory row n (fixed)

    v8f acc[4];
    v8f zero = {0.f,0.f,0.f,0.f,0.f,0.f,0.f,0.f};
    #pragma unroll
    for (int mt = 0; mt < 4; ++mt) acc[mt] = zero;

    wmma_tile(ldsA, (const __bf16*)SNu, srow, nsub, half, acc);

    int mbase = 8 * half;
    #pragma unroll
    for (int mt = 0; mt < 4; ++mt) {
        size_t off = (size_t)srow * NUM_DRIVES + mt * 16 + mbase;  // 8 contig q
        v8f pold;
        if (t > 1) pold = *(const v8f*)(penalty + off);
        v8f svec = *(const v8f*)(simsT + off);
        v8f pnew;
        #pragma unroll
        for (int r = 0; r < 8; ++r) {
            int q = mt * 16 + mbase + r;
            float p = acc[mt][r];
            if (t > 1) p = fmaxf(pold[r], p);
            pnew[r] = p;
            float s = svec[r];
            float adj = ((s >= THRESHOLD) ? s : NEGV) - DIV_LAMBDA * p;
            v4i sl = ssel4[q];
            if (srow == sl[0] || srow == sl[1] || srow == sl[2] || srow == sl[3])
                adj = NEGV;
            u64 pk = ((u64)float_key(adj) << 32) | (u32)(~(u32)srow);
            // per-q max goes straight to LDS (each lane's q's are distinct)
            atomicMax(&sbest[q], pk);
        }
        if (t < NUM_RESULTS - 1)             // penalty dead after step 4
            *(v8f*)(penalty + off) = pnew;
    }
    __syncthreads();
    if (tid < NUM_DRIVES) atomicMax(&slots[tid], sbest[tid]);
}

// =====================================================================
// 5) Decode argmax, record index, gather selected normalized bf16 rows,
//    reset slots for next step.  One block of 256 threads.
// =====================================================================
__global__ __launch_bounds__(256) void select_kernel(
    u64* __restrict__ slots, int* __restrict__ selected,
    const unsigned short* __restrict__ snu, unsigned short* __restrict__ selu, int t)
{
    __shared__ int sidx[NUM_DRIVES];
    int tid = threadIdx.x;
    if (tid < NUM_DRIVES) {
        u64 v = slots[tid];
        int idx = (int)(~(u32)(v & 0xffffffffull));
        sidx[tid] = idx;
        selected[t * 64 + tid] = idx;
        slots[tid] = 0ull;
    }
    __syncthreads();

    int q = tid >> 2;
    int p = tid & 3;
    int idx = sidx[q];
    const __bf16* src = (const __bf16*)snu + (size_t)idx * SENSORY_DIM + p * 64;
    __bf16*       dst = (__bf16*)selu      + (size_t)q   * SENSORY_DIM + p * 64;
    #pragma unroll
    for (int i = 0; i < 64; i += 8)
        *(v8bf*)(dst + i) = *(const v8bf*)(src + i);
}

// =====================================================================
// 6) init + finalize
// =====================================================================
__global__ void init_kernel(u64* slots, int* counts)
{
    int t = threadIdx.x;
    if (t < NUM_DRIVES) { slots[t] = 0ull; counts[t] = 0; }
}

__global__ __launch_bounds__(256) void finalize_kernel(
    const int* __restrict__ selected, const int* __restrict__ counts,
    const float* __restrict__ mem_loc, const float* __restrict__ mem_sds,
    const float* __restrict__ mem_senses, float* __restrict__ out)
{
    int q = blockIdx.x;
    int tid = threadIdx.x;

    if (tid < NUM_RESULTS * LOCATION_DIM) {           // 80
        int j = tid / LOCATION_DIM, d = tid % LOCATION_DIM;
        int idx = selected[j * 64 + q];
        out[OUT_LOC + ((size_t)(q * NUM_RESULTS + j)) * LOCATION_DIM + d] =
            mem_loc[(size_t)idx * LOCATION_DIM + d];
        out[OUT_SDS + ((size_t)(q * NUM_RESULTS + j)) * LOCATION_DIM + d] =
            mem_sds[(size_t)idx * LOCATION_DIM + d];
    }
    for (int e = tid; e < NUM_RESULTS * SENSORY_DIM; e += 256) {  // 1280
        int j = e / SENSORY_DIM, d = e % SENSORY_DIM;
        int idx = selected[j * 64 + q];
        out[OUT_SENSE + ((size_t)(q * NUM_RESULTS + j)) * SENSORY_DIM + d] =
            mem_senses[(size_t)idx * SENSORY_DIM + d];
    }
    if (tid == 0) {
        int c = counts[q]; if (c > NUM_RESULTS) c = NUM_RESULTS;
        ((int*)out)[OUT_NFOUND + q] = c;   // int32 bits into output buffer
    }
}

// =====================================================================
// launch
// =====================================================================
extern "C" void kernel_launch(void* const* d_in, const int* in_sizes, int n_in,
                              void* d_out, int out_size, void* d_ws, size_t ws_size,
                              hipStream_t stream)
{
    const float* drive_weights    = (const float*)d_in[0];  // 64x256
    const float* mem_locations    = (const float*)d_in[1];  // 200000x16
    const float* mem_location_sds = (const float*)d_in[2];  // 200000x16
    const float* mem_senses       = (const float*)d_in[3];  // 200000x256

    char* ws = (char*)d_ws;
    unsigned short* sn   = (unsigned short*)(ws + WS_SN);
    unsigned short* qn   = (unsigned short*)(ws + WS_QN);
    unsigned short* selb = (unsigned short*)(ws + WS_SEL);
    float*          simsT   = (float*)(ws + WS_SIMS);
    float*          penalty = (float*)(ws + WS_PEN);
    u64*            slots   = (u64*)(ws + WS_SLOTS);
    int*            counts  = (int*)(ws + WS_COUNTS);
    int*            selidx  = (int*)(ws + WS_SELIDX);

    normalize_bf16_kernel<<<MEM_SIZE / 8, 256, 0, stream>>>(mem_senses, sn, MEM_SIZE);
    normalize_bf16_kernel<<<NUM_DRIVES / 8, 256, 0, stream>>>(drive_weights, qn, NUM_DRIVES);
    init_kernel<<<1, 64, 0, stream>>>(slots, counts);

    // sims = qn . sn^T  (bf16 WMMA, A in LDS)
    gemm_sims_kernel<<<GEMM_BLOCKS, 128, 0, stream>>>(qn, sn, simsT);

    // step 0: argmax over masked sims + valid count
    const int passBlocks = (MEM_SIZE + 127) / 128;
    pass0_kernel<<<passBlocks, 256, 0, stream>>>(simsT, slots, counts);
    select_kernel<<<1, 256, 0, stream>>>(slots, selidx, sn, selb, 0);

    // steps 1..4: fused WMMA penalty update + argmax
    for (int t = 1; t < NUM_RESULTS; ++t) {
        fused_step_kernel<<<GEMM_BLOCKS, 128, 0, stream>>>(selb, sn, penalty,
                                                           simsT, selidx, slots, t);
        select_kernel<<<1, 256, 0, stream>>>(slots, selidx, sn, selb, t);
    }

    finalize_kernel<<<NUM_DRIVES, 256, 0, stream>>>(selidx, counts, mem_locations,
                                                    mem_location_sds, mem_senses,
                                                    (float*)d_out);
}
